// LocalDino_67808943669757
// MI455X (gfx1250) — matrix operand order repaired and checked by reading, not compiled, not run on hardware
//
#include <hip/hip_runtime.h>
#include <hip/hip_bf16.h>
#include <math.h>

typedef __bf16 bf16;
typedef __attribute__((ext_vector_type(16))) __bf16 v16bf;
typedef __attribute__((ext_vector_type(8)))  __bf16 bf16x8;
typedef __attribute__((ext_vector_type(8)))  float  v8f;

#define DEV __device__ __forceinline__

// ---------------------------------------------------------------------------
// WMMA helpers (CDNA5 gfx1250, wave32).
// A frag (16x32 bf16): lane holds row M=lane&15; half=lane>>4.
//   elements 0..7  -> K = half*8 + {0..7}        (contiguous, one b128 load)
//   elements 8..15 -> K = 16 + half*8 + {0..7}   (contiguous, one b128 load)
// B frag (32x16 bf16, operand stored row-major [N,K]): lane holds col N=lane&15;
//   elements 0..7 -> K = half*8 + {0..7}, 8..15 -> K = 16 + half*8 + {0..7} (two b128 loads)
// C/D f32: lane holds col N=lane&15; acc[v] is row M = v + 8*(lane>>4).
// ---------------------------------------------------------------------------
DEV v16bf pack16(bf16x8 lo, bf16x8 hi) {
  v16bf f;
#pragma unroll
  for (int i = 0; i < 8; i++) { f[i] = lo[i]; f[i + 8] = hi[i]; }
  return f;
}
DEV v16bf ldA(const bf16* p) { return pack16(*(const bf16x8*)p, *(const bf16x8*)(p + 16)); }
DEV v16bf ldB(const bf16* p) { return pack16(*(const bf16x8*)p, *(const bf16x8*)(p + 16)); }
DEV v8f wmma_bf16(v16bf a, v16bf b, v8f c) {
  return __builtin_amdgcn_wmma_f32_16x16x32_bf16(false, a, false, b, (short)0, c, false, false);
}

DEV void store_tile(v8f acc, float* C, bf16* Cb, size_t cbase, int ldc,
                    int m, int n, const float* bias, int epi) {
  float bv = bias ? bias[n] : 0.0f;
#pragma unroll
  for (int v = 0; v < 8; v++) {
    float x = acc[v] + bv;
    if (epi == 1) x = 0.5f * x * (1.0f + erff(x * 0.70710678118654752f));  // exact GELU
    size_t o = cbase + (size_t)(m + v) * ldc + n;
    if (C)  C[o]  = x;
    if (Cb) Cb[o] = (bf16)x;
  }
}

// ---------------------------------------------------------------------------
// Generic batched GEMM: C[M,N] = act(A[M,K] @ B[N,K]^T + bias)
// grid = (M/128, N/64, batch); block = 256 (8 waves, each a 32x32 tile).
// epi: 0 = bias only, 1 = bias + exact GELU.
// ---------------------------------------------------------------------------
__global__ __launch_bounds__(256) void k_gemm(
    const bf16* __restrict__ A, const bf16* __restrict__ B,
    const float* __restrict__ bias, float* __restrict__ C, bf16* __restrict__ Cb,
    int K, int lda, int ldb, int ldc,
    long long sA, long long sB, long long sC, int epi) {
  const int lane = threadIdx.x & 31;
  const int wave = threadIdx.x >> 5;
  const int m0 = blockIdx.x * 128 + (wave & 3) * 32;
  const int n0 = blockIdx.y * 64 + (wave >> 2) * 32;
  const int row = lane & 15, half = lane >> 4;

  const bf16* Ab = A + (size_t)blockIdx.z * sA;
  const bf16* Bb = B + (size_t)blockIdx.z * sB;
  const size_t cbase = (size_t)blockIdx.z * sC;

  const bf16* a0p = Ab + (size_t)(m0 + row) * lda + half * 8;
  const bf16* a1p = a0p + (size_t)16 * lda;
  const bf16* b0p = Bb + (size_t)(n0 + row) * ldb + half * 8;
  const bf16* b1p = b0p + (size_t)16 * ldb;

  v8f acc00 = {}, acc01 = {}, acc10 = {}, acc11 = {};
  for (int kb = 0; kb < K; kb += 32) {
    v16bf a0 = ldA(a0p + kb), a1 = ldA(a1p + kb);
    v16bf b0 = ldB(b0p + kb), b1 = ldB(b1p + kb);
    acc00 = wmma_bf16(a0, b0, acc00);
    acc01 = wmma_bf16(a0, b1, acc01);
    acc10 = wmma_bf16(a1, b0, acc10);
    acc11 = wmma_bf16(a1, b1, acc11);
  }
  int mrow = m0 + 8 * half, ncol = n0 + row;
  store_tile(acc00, C, Cb, cbase, ldc, mrow,      ncol,      bias, epi);
  store_tile(acc01, C, Cb, cbase, ldc, mrow,      ncol + 16, bias, epi);
  store_tile(acc10, C, Cb, cbase, ldc, mrow + 16, ncol,      bias, epi);
  store_tile(acc11, C, Cb, cbase, ldc, mrow + 16, ncol + 16, bias, epi);
}

// ---------------------------------------------------------------------------
// Implicit-im2col 3x3 SAME conv as WMMA GEMM.
// M = imgs*H*W (pixel-major rows), N = Co, K = Ci*9 ordered (ci,ky,kx) to match
// the row-major weight layout [Co, Ci, 3, 3] -> B[N,K].
// Epilogue: relu(acc*scale[n] + shift[n])  (conv-bias + BN folded).
// Output: f32 [M, Co] pixel-major.
// ---------------------------------------------------------------------------
DEV v16bf ld_conv_frag(const bf16* __restrict__ X, int Ci, int H, int W,
                       int img, int y, int x, int kb, int half) {
  v16bf f;
#pragma unroll
  for (int e = 0; e < 16; e++) {
    int k = kb + ((e < 8) ? half * 8 + e : 16 + half * 8 + (e - 8));
    int ci = k / 9; int t = k - ci * 9; int ky = t / 3; int kx = t - ky * 3;
    int iy = y + ky - 1, ix = x + kx - 1;
    bf16 v = (bf16)0.0f;
    if ((unsigned)iy < (unsigned)H && (unsigned)ix < (unsigned)W)
      v = X[(((size_t)img * Ci + ci) * H + iy) * W + ix];
    f[e] = v;
  }
  return f;
}

__global__ __launch_bounds__(256) void k_convgemm(
    const bf16* __restrict__ X, const bf16* __restrict__ Wt,
    const float* __restrict__ scale, const float* __restrict__ shift,
    float* __restrict__ C, int Ci, int H, int W, int Co, int Kt) {
  const int lane = threadIdx.x & 31;
  const int wave = threadIdx.x >> 5;
  const int m0 = blockIdx.x * 128 + (wave & 3) * 32;
  const int n0 = blockIdx.y * 64 + (wave >> 2) * 32;
  const int row = lane & 15, half = lane >> 4;
  const int HW = H * W;

  int r0 = m0 + row, r1 = r0 + 16;
  int img0 = r0 / HW, p0 = r0 - img0 * HW, y0 = p0 / W, x0 = p0 - y0 * W;
  int img1 = r1 / HW, p1 = r1 - img1 * HW, y1 = p1 / W, x1 = p1 - y1 * W;

  const bf16* b0p = Wt + (size_t)(n0 + row) * Kt + half * 8;
  const bf16* b1p = b0p + (size_t)16 * Kt;

  v8f acc00 = {}, acc01 = {}, acc10 = {}, acc11 = {};
  for (int kb = 0; kb < Kt; kb += 32) {
    v16bf a0 = ld_conv_frag(X, Ci, H, W, img0, y0, x0, kb, half);
    v16bf a1 = ld_conv_frag(X, Ci, H, W, img1, y1, x1, kb, half);
    v16bf b0 = ldB(b0p + kb);
    v16bf b1 = ldB(b1p + kb);
    acc00 = wmma_bf16(a0, b0, acc00);
    acc01 = wmma_bf16(a0, b1, acc01);
    acc10 = wmma_bf16(a1, b0, acc10);
    acc11 = wmma_bf16(a1, b1, acc11);
  }
  int mrow = m0 + 8 * half, ncol = n0 + row;
#pragma unroll
  for (int q = 0; q < 4; q++) {
    v8f acc = (q == 0) ? acc00 : (q == 1) ? acc01 : (q == 2) ? acc10 : acc11;
    int m = mrow + ((q >= 2) ? 16 : 0);
    int n = ncol + ((q & 1) ? 16 : 0);
    float sc = scale[n], sh = shift[n];
#pragma unroll
    for (int v = 0; v < 8; v++) {
      float x = acc[v] * sc + sh;
      C[(size_t)(m + v) * Co + n] = x > 0.0f ? x : 0.0f;
    }
  }
}

// --------------------------- elementwise glue ------------------------------
__global__ void k_f2b(const float* __restrict__ x, bf16* __restrict__ y, size_t n) {
  size_t i = (size_t)blockIdx.x * blockDim.x + threadIdx.x;
  size_t st = (size_t)gridDim.x * blockDim.x;
  for (; i < n; i += st) y[i] = (bf16)x[i];
}

// im2col for the 16x16/stride-16 patch embed: A[4096 tokens, 768]
__global__ void k_im2col(const float* __restrict__ patches, bf16* __restrict__ A) {
  size_t idx = (size_t)blockIdx.x * blockDim.x + threadIdx.x;
  if (idx >= (size_t)4096 * 768) return;
  int t = (int)(idx / 768), c = (int)(idx % 768);
  int img = t >> 6, tok = t & 63, gy = tok >> 3, gx = tok & 7;
  int ch = c >> 8, r = c & 255, py = r >> 4, px = r & 15;
  A[idx] = (bf16)patches[(((size_t)img * 3 + ch) * 128 + (gy * 16 + py)) * 128 + (gx * 16 + px)];
}

// qkv f32 [4096, 3072] -> q (scaled 1/sqrt(128)) / k as [bh, 512, 128] bf16,
// v transposed to [bh, 128, 512] bf16 so P@V uses the [N,K] B layout.
__global__ void k_extract(const float* __restrict__ qkv, bf16* __restrict__ q,
                          bf16* __restrict__ k, bf16* __restrict__ vt) {
  size_t idx = (size_t)blockIdx.x * blockDim.x + threadIdx.x;
  if (idx >= (size_t)4096 * 1024) return;
  int t = (int)(idx >> 10), c = (int)(idx & 1023);
  int b = t >> 9, s = t & 511, h = c >> 7, d = c & 127;
  const float* row = qkv + (size_t)t * 3072;
  size_t bh = (size_t)(b * 8 + h);
  q[(bh * 512 + s) * 128 + d]  = (bf16)(row[c] * 0.08838834764831845f);
  k[(bh * 512 + s) * 128 + d]  = (bf16)row[1024 + c];
  vt[(bh * 128 + d) * 512 + s] = (bf16)row[2048 + c];
}

// one wave per row of 512; wave32 shfl reductions; write bf16 probabilities
__global__ __launch_bounds__(256) void k_softmax(const float* __restrict__ S,
                                                 bf16* __restrict__ P) {
  int lane = threadIdx.x & 31;
  int row = blockIdx.x * 8 + (threadIdx.x >> 5);
  const float* r = S + (size_t)row * 512;
  float v[16], mx = -1e30f;
#pragma unroll
  for (int j = 0; j < 16; j++) { v[j] = r[lane + 32 * j]; mx = fmaxf(mx, v[j]); }
  for (int o = 16; o > 0; o >>= 1) mx = fmaxf(mx, __shfl_xor(mx, o, 32));
  float sum = 0.0f;
#pragma unroll
  for (int j = 0; j < 16; j++) { v[j] = expf(v[j] - mx); sum += v[j]; }
  for (int o = 16; o > 0; o >>= 1) sum += __shfl_xor(sum, o, 32);
  float inv = 1.0f / sum;
  bf16* pr = P + (size_t)row * 512;
#pragma unroll
  for (int j = 0; j < 16; j++) pr[lane + 32 * j] = (bf16)(v[j] * inv);
}

// merge heads: O [bh,512,128] f32 -> ctx [4096,1024] bf16
__global__ void k_merge(const float* __restrict__ O, bf16* __restrict__ ctx) {
  size_t idx = (size_t)blockIdx.x * blockDim.x + threadIdx.x;
  if (idx >= (size_t)4096 * 1024) return;
  int t = (int)(idx >> 10), c = (int)(idx & 1023);
  int b = t >> 9, s = t & 511, h = c >> 7, d = c & 127;
  ctx[idx] = (bf16)O[((size_t)(b * 8 + h) * 512 + s) * 128 + d];
}

// hs = LN(resid + sub); one 256-thread block per token (D=1024)
__global__ __launch_bounds__(256) void k_resln(
    const float* __restrict__ resid, const float* __restrict__ sub,
    const float* __restrict__ g, const float* __restrict__ b,
    float* __restrict__ out, bf16* __restrict__ outb) {
  __shared__ float red[256];
  int t = blockIdx.x, tx = threadIdx.x;
  const float* xr = resid + (size_t)t * 1024;
  const float* sr = sub + (size_t)t * 1024;
  float v[4], s = 0.0f;
#pragma unroll
  for (int i = 0; i < 4; i++) { v[i] = xr[tx + 256 * i] + sr[tx + 256 * i]; s += v[i]; }
  red[tx] = s; __syncthreads();
  for (int st = 128; st > 0; st >>= 1) { if (tx < st) red[tx] += red[tx + st]; __syncthreads(); }
  float mean = red[0] * (1.0f / 1024.0f); __syncthreads();
  float sq = 0.0f;
#pragma unroll
  for (int i = 0; i < 4; i++) { float d = v[i] - mean; sq += d * d; }
  red[tx] = sq; __syncthreads();
  for (int st = 128; st > 0; st >>= 1) { if (tx < st) red[tx] += red[tx + st]; __syncthreads(); }
  float inv = rsqrtf(red[0] * (1.0f / 1024.0f) + 1e-5f);
#pragma unroll
  for (int i = 0; i < 4; i++) {
    int c = tx + 256 * i;
    float y = (v[i] - mean) * inv * g[c] + b[c];
    out[(size_t)t * 1024 + c] = y;
    outb[(size_t)t * 1024 + c] = (bf16)y;
  }
}

// tokens [img*64+pix, 1024] f32 -> NCHW bf16 [64,1024,8,8]
__global__ void k_tok2nchw(const float* __restrict__ hs, bf16* __restrict__ fm) {
  size_t idx = (size_t)blockIdx.x * blockDim.x + threadIdx.x;
  if (idx >= (size_t)64 * 1024 * 64) return;
  int img = (int)(idx >> 16), rem = (int)(idx & 65535);
  int d = rem >> 6, p = rem & 63;
  fm[idx] = (bf16)hs[((size_t)img * 64 + p) * 1024 + d];
}

// fold conv bias + eval BN into per-channel scale/shift
__global__ void k_bnfold(const float* db, const float* g, const float* be,
                         const float* m, const float* v,
                         float* scale, float* shift, int co) {
  int i = blockIdx.x * blockDim.x + threadIdx.x;
  if (i >= co) return;
  float s = g[i] * rsqrtf(v[i] + 1e-5f);
  scale[i] = s;
  shift[i] = (db[i] - m[i]) * s + be[i];
}

// bilinear x2 (half-pixel) from pixel-major f32 [img*H*W, C] to NCHW bf16
__global__ void k_up2(const float* __restrict__ cin, bf16* __restrict__ out,
                      int C, int H, int W) {
  int OH = 2 * H, OW = 2 * W;
  size_t total = (size_t)64 * C * OH * OW;
  size_t idx = (size_t)blockIdx.x * blockDim.x + threadIdx.x;
  if (idx >= total) return;
  int ox = (int)(idx % OW);
  size_t r = idx / OW;
  int oy = (int)(r % OH); r /= OH;
  int c = (int)(r % C);
  int img = (int)(r / C);
  float sy = (oy + 0.5f) * 0.5f - 0.5f;
  float sx = (ox + 0.5f) * 0.5f - 0.5f;
  int yy0 = (int)floorf(sy), xx0 = (int)floorf(sx);
  float fy = sy - yy0, fx = sx - xx0;
  int yl = min(max(yy0, 0), H - 1), yh = min(max(yy0 + 1, 0), H - 1);
  int xl = min(max(xx0, 0), W - 1), xh = min(max(xx0 + 1, 0), W - 1);
  size_t base = (size_t)img * H * W;
  float v00 = cin[(base + (size_t)yl * W + xl) * C + c];
  float v01 = cin[(base + (size_t)yl * W + xh) * C + c];
  float v10 = cin[(base + (size_t)yh * W + xl) * C + c];
  float v11 = cin[(base + (size_t)yh * W + xh) * C + c];
  float v0 = v00 + (v01 - v00) * fx;
  float v1 = v10 + (v11 - v10) * fx;
  out[idx] = (bf16)(v0 + (v1 - v0) * fy);
}

// final 1x1 conv: 64 channels -> 1, f32 output straight into d_out
__global__ void k_conv1x1(const bf16* __restrict__ x, const float* __restrict__ w,
                          const float* __restrict__ b, float* __restrict__ out) {
  size_t idx = (size_t)blockIdx.x * blockDim.x + threadIdx.x;
  if (idx >= (size_t)64 * 16384) return;
  int img = (int)(idx >> 14), p = (int)(idx & 16383);
  float acc = b[0];
#pragma unroll 8
  for (int c = 0; c < 64; c++)
    acc += (float)x[((size_t)img * 64 + c) * 16384 + p] * w[c];
  out[idx] = acc;
}

// ---------------------------------------------------------------------------
extern "C" void kernel_launch(void* const* d_in, const int* in_sizes, int n_in,
                              void* d_out, int out_size, void* d_ws, size_t ws_size,
                              hipStream_t stream) {
  (void)in_sizes; (void)n_in; (void)out_size; (void)ws_size;
  const float* pe_w  = (const float*)d_in[1];
  const float* pe_b  = (const float*)d_in[2];
  const float* qkv_w = (const float*)d_in[3];
  const float* qkv_b = (const float*)d_in[4];
  const float* out_w = (const float*)d_in[5];
  const float* out_b = (const float*)d_in[6];
  const float* ln1_g = (const float*)d_in[7];
  const float* ln1_b = (const float*)d_in[8];
  const float* ff1_w = (const float*)d_in[9];
  const float* ff1_b = (const float*)d_in[10];
  const float* ff2_w = (const float*)d_in[11];
  const float* ff2_b = (const float*)d_in[12];
  const float* ln2_g = (const float*)d_in[13];
  const float* ln2_b = (const float*)d_in[14];
  const float *dw[5], *db[5], *gg[4], *be[4], *mm[4], *vv[4];
  for (int i = 0; i < 4; i++) {
    int base = 15 + i * 6;
    dw[i] = (const float*)d_in[base];
    db[i] = (const float*)d_in[base + 1];
    gg[i] = (const float*)d_in[base + 2];
    be[i] = (const float*)d_in[base + 3];
    mm[i] = (const float*)d_in[base + 4];
    vv[i] = (const float*)d_in[base + 5];
  }
  dw[4] = (const float*)d_in[39];
  db[4] = (const float*)d_in[40];

  // bump allocator over d_ws
  char* w = (char*)d_ws; size_t off = 0;
  auto alloc = [&](size_t bytes) -> void* {
    void* p = w + off; off = (off + bytes + 255) & ~(size_t)255; return p;
  };
  bf16* wpe  = (bf16*)alloc((size_t)1024 * 768 * 2);
  bf16* wqkv = (bf16*)alloc((size_t)2 * 3072 * 1024 * 2);
  bf16* wout = (bf16*)alloc((size_t)2 * 1024 * 1024 * 2);
  bf16* wff1 = (bf16*)alloc((size_t)2 * 4096 * 1024 * 2);
  bf16* wff2 = (bf16*)alloc((size_t)2 * 1024 * 4096 * 2);
  bf16* wc1  = (bf16*)alloc((size_t)512 * 9216 * 2);
  bf16* wc2  = (bf16*)alloc((size_t)256 * 4608 * 2);
  bf16* wc3  = (bf16*)alloc((size_t)128 * 2304 * 2);
  bf16* wc4  = (bf16*)alloc((size_t)64 * 1152 * 2);
  bf16* Apb  = (bf16*)alloc((size_t)4096 * 768 * 2);
  float* hs  = (float*)alloc((size_t)4096 * 1024 * 4);
  bf16* hsb  = (bf16*)alloc((size_t)4096 * 1024 * 2);
  float* qkvbuf = (float*)alloc((size_t)4096 * 3072 * 4);
  bf16* qh  = (bf16*)alloc((size_t)64 * 512 * 128 * 2);
  bf16* kh  = (bf16*)alloc((size_t)64 * 512 * 128 * 2);
  bf16* vth = (bf16*)alloc((size_t)64 * 128 * 512 * 2);
  float* sc = (float*)alloc((size_t)64 * 512 * 512 * 4);
  bf16* pr  = (bf16*)alloc((size_t)64 * 512 * 512 * 2);
  float* Ob = (float*)alloc((size_t)64 * 512 * 128 * 4);
  bf16* ctx = (bf16*)alloc((size_t)4096 * 1024 * 2);
  float* sub = (float*)alloc((size_t)4096 * 1024 * 4);  // proj / ff2 output
  bf16* ff1b = (bf16*)alloc((size_t)4096 * 4096 * 2);
  float* ssc = (float*)alloc((size_t)512 * 4);
  float* ssh = (float*)alloc((size_t)512 * 4);
  bf16* x1  = (bf16*)alloc((size_t)64 * 1024 * 64 * 2);
  float* c1o = (float*)alloc((size_t)4096 * 512 * 4);
  bf16* x2  = (bf16*)alloc((size_t)64 * 512 * 256 * 2);
  float* c2o = (float*)alloc((size_t)16384 * 256 * 4);
  bf16* x3  = (bf16*)alloc((size_t)64 * 256 * 1024 * 2);
  float* c3o = (float*)alloc((size_t)65536 * 128 * 4);
  bf16* x4  = (bf16*)alloc((size_t)64 * 128 * 4096 * 2);
  float* c4o = (float*)alloc((size_t)262144 * 64 * 4);
  bf16* x5  = (bf16*)alloc((size_t)64 * 64 * 16384 * 2);

  auto cvt = [&](const float* src, bf16* dst, size_t n) {
    int blocks = (int)((n + 255) / 256); if (blocks > 65535) blocks = 65535;
    k_f2b<<<blocks, 256, 0, stream>>>(src, dst, n);
  };
  auto gemm = [&](const bf16* A, const bf16* B, const float* bias, float* C, bf16* Cb,
                  int M, int N, int K, int lda, int ldb, int ldc,
                  long long sA, long long sB, long long sC, int batch, int epi) {
    dim3 g(M / 128, N / 64, batch);
    k_gemm<<<g, 256, 0, stream>>>(A, B, bias, C, Cb, K, lda, ldb, ldc, sA, sB, sC, epi);
  };

  // ---- weight conversion to bf16 (L2-resident operands for WMMA) ----
  cvt(pe_w, wpe, (size_t)1024 * 768);
  cvt(qkv_w, wqkv, (size_t)2 * 3072 * 1024);
  cvt(out_w, wout, (size_t)2 * 1024 * 1024);
  cvt(ff1_w, wff1, (size_t)2 * 4096 * 1024);
  cvt(ff2_w, wff2, (size_t)2 * 1024 * 4096);
  cvt(dw[0], wc1, (size_t)512 * 9216);
  cvt(dw[1], wc2, (size_t)256 * 4608);
  cvt(dw[2], wc3, (size_t)128 * 2304);
  cvt(dw[3], wc4, (size_t)64 * 1152);

  // ---- patch embed: im2col + WMMA GEMM [4096,768]x[1024,768]^T ----
  k_im2col<<<(int)(((size_t)4096 * 768 + 255) / 256), 256, 0, stream>>>(
      (const float*)d_in[0], Apb);
  gemm(Apb, wpe, pe_b, hs, hsb, 4096, 1024, 768, 768, 768, 1024, 0, 0, 0, 1, 0);

  // ---- transformer encoder (post-norm, exact GELU) ----
  const int NE = (int)(((size_t)4096 * 1024 + 255) / 256);
  for (int i = 0; i < 2; i++) {
    gemm(hsb, wqkv + (size_t)i * 3072 * 1024, qkv_b + (size_t)i * 3072,
         qkvbuf, nullptr, 4096, 3072, 1024, 1024, 1024, 3072, 0, 0, 0, 1, 0);
    k_extract<<<NE, 256, 0, stream>>>(qkvbuf, qh, kh, vth);
    // scores: per (b,h) 512x512 = Q(512x128) @ K(512x128)^T, batched over 64
    gemm(qh, kh, nullptr, sc, nullptr, 512, 512, 128, 128, 128, 512,
         512 * 128, 512 * 128, 512 * 512, 64, 0);
    k_softmax<<<4096, 256, 0, stream>>>(sc, pr);
    // O = P(512x512) @ V ; V pre-transposed to [128,512] -> [N,K] operand
    gemm(pr, vth, nullptr, Ob, nullptr, 512, 128, 512, 512, 512, 128,
         512 * 512, 128 * 512, 512 * 128, 64, 0);
    k_merge<<<NE, 256, 0, stream>>>(Ob, ctx);
    gemm(ctx, wout + (size_t)i * 1024 * 1024, out_b + (size_t)i * 1024,
         sub, nullptr, 4096, 1024, 1024, 1024, 1024, 1024, 0, 0, 0, 1, 0);
    k_resln<<<4096, 256, 0, stream>>>(hs, sub, ln1_g + i * 1024, ln1_b + i * 1024, hs, hsb);
    gemm(hsb, wff1 + (size_t)i * 4096 * 1024, ff1_b + (size_t)i * 4096,
         nullptr, ff1b, 4096, 4096, 1024, 1024, 1024, 4096, 0, 0, 0, 1, 1);
    gemm(ff1b, wff2 + (size_t)i * 1024 * 4096, ff2_b + (size_t)i * 1024,
         sub, nullptr, 4096, 1024, 4096, 4096, 4096, 1024, 0, 0, 0, 1, 0);
    k_resln<<<4096, 256, 0, stream>>>(hs, sub, ln2_g + i * 1024, ln2_b + i * 1024, hs, hsb);
  }

  // ---- CNN decoder: 4x (implicit-im2col WMMA conv + fused BN/ReLU + up2) ----
  k_tok2nchw<<<(int)(((size_t)64 * 1024 * 64 + 255) / 256), 256, 0, stream>>>(hs, x1);

  k_bnfold<<<2, 256, 0, stream>>>(db[0], gg[0], be[0], mm[0], vv[0], ssc, ssh, 512);
  k_convgemm<<<dim3(32, 8), 256, 0, stream>>>(x1, wc1, ssc, ssh, c1o, 1024, 8, 8, 512, 9216);
  k_up2<<<(int)(((size_t)64 * 512 * 16 * 16 + 255) / 256), 256, 0, stream>>>(c1o, x2, 512, 8, 8);

  k_bnfold<<<1, 256, 0, stream>>>(db[1], gg[1], be[1], mm[1], vv[1], ssc, ssh, 256);
  k_convgemm<<<dim3(128, 4), 256, 0, stream>>>(x2, wc2, ssc, ssh, c2o, 512, 16, 16, 256, 4608);
  k_up2<<<(int)(((size_t)64 * 256 * 32 * 32 + 255) / 256), 256, 0, stream>>>(c2o, x3, 256, 16, 16);

  k_bnfold<<<1, 128, 0, stream>>>(db[2], gg[2], be[2], mm[2], vv[2], ssc, ssh, 128);
  k_convgemm<<<dim3(512, 2), 256, 0, stream>>>(x3, wc3, ssc, ssh, c3o, 256, 32, 32, 128, 2304);
  k_up2<<<(int)(((size_t)64 * 128 * 64 * 64 + 255) / 256), 256, 0, stream>>>(c3o, x4, 128, 32, 32);

  k_bnfold<<<1, 64, 0, stream>>>(db[3], gg[3], be[3], mm[3], vv[3], ssc, ssh, 64);
  k_convgemm<<<dim3(2048, 1), 256, 0, stream>>>(x4, wc4, ssc, ssh, c4o, 128, 64, 64, 64, 1152);
  k_up2<<<(int)(((size_t)64 * 64 * 128 * 128 + 255) / 256), 256, 0, stream>>>(c4o, x5, 64, 64, 64);

  k_conv1x1<<<(int)(((size_t)64 * 16384 + 255) / 256), 256, 0, stream>>>(
      x5, dw[4], db[4], (float*)d_out);
}
